// LSTMSupertaggingModel_35759897706657
// MI455X (gfx1250) — matrix operand order repaired and checked by reading, not compiled, not run on hardware
//
#include <hip/hip_runtime.h>
#include <hip/hip_bf16.h>

typedef _Float16 h16;
typedef __attribute__((ext_vector_type(16))) _Float16 v16h;
typedef __attribute__((ext_vector_type(8)))  float    v8f;
typedef __attribute__((ext_vector_type(4)))  unsigned int u32x4;
typedef __attribute__((ext_vector_type(8)))  unsigned int u32x8;

// Model dims
#define BB 8
#define SS 512
#define DD 768
#define NHD 12
#define DHD 64
#define FF 3072
#define LL 12
#define LH 384
#define G4 1536   // 4*LH
#define NC 425
#define NCP 448   // NC padded to 64
#define WW 128
#define ROWS (BB*SS) // 4096

// ---------------------------------------------------------------------------
// WMMA helpers (CDNA5 16x16x32 f16 -> f32)
// ---------------------------------------------------------------------------
__device__ __forceinline__ v8f wmma32(v16h a, v16h b, v8f c) {
    return __builtin_amdgcn_wmma_f32_16x16x32_f16(false, a, false, b, (short)0, c,
                                                  false, false);
}
__device__ __forceinline__ v8f zero8() {
    v8f z;
#pragma unroll
    for (int i = 0; i < 8; ++i) z[i] = 0.0f;
    return z;
}
__device__ __forceinline__ v16h ldv(const h16* p) { return *(const v16h*)(p); }

// A-fragment: 16x32 f16 tile, row-major with leading dim ld.
// lanes hold M=lane&15; lane-half selects K subset {kb..kb+15}, kb = 16*half
__device__ __forceinline__ v16h frag_a(const h16* base, int ld, int lane) {
    int r = lane & 15;
    int kb = (lane >> 4) * 16;
    const h16* p = base + (size_t)r * ld;
    v16h f;
#pragma unroll
    for (int j = 0; j < 8; ++j) f[j] = p[kb + j];
#pragma unroll
    for (int j = 0; j < 8; ++j) f[8 + j] = p[kb + 8 + j];
    return f;
}

// Swizzled-B tile index: within a 32(K)x64(N) tile, element (k,n) lives at
//   (n/16)*512 + (((k/16)&1)*16 + (n&15))*16 + (k&15)
// so a B-fragment (c, lane) is 16 contiguous halves at  c*512 + lane*16.
__device__ __forceinline__ size_t swB_idx(int k, int nn) {
    return (size_t)((nn & 63) >> 4) * 512 +
           (size_t)((((k >> 4) & 1) << 4) + (nn & 15)) * 16 + (k & 15);
}

__device__ __forceinline__ float sigf(float x) { return 1.0f / (1.0f + expf(-x)); }

// low 32 bits of a flat pointer into the LDS aperture == LDS byte offset
__device__ __forceinline__ unsigned int lds_off(const void* p) {
    return (unsigned int)(size_t)p;
}

// ---------------------------------------------------------------------------
// Weight conversion: f32 [K][N] row-major -> f16 fragment-swizzled tiles.
// Tiles ordered  (k/32) * (Np/64) + (n/64), 2048 halves each. Pads N -> Np.
// ---------------------------------------------------------------------------
__global__ void k_cvtB(const float* __restrict__ s, h16* __restrict__ d,
                       int K, int N, int Np) {
    size_t tot = (size_t)K * Np;
    int nt64 = Np >> 6;
    for (size_t i = (size_t)blockIdx.x * blockDim.x + threadIdx.x; i < tot;
         i += (size_t)gridDim.x * blockDim.x) {
        int k = (int)(i / Np), nn = (int)(i % Np);
        float v = (nn < N) ? s[(size_t)k * N + nn] : 0.0f;
        size_t base = ((size_t)(k >> 5) * nt64 + (nn >> 6)) * 2048;
        d[base + swB_idx(k, nn)] = (h16)v;
    }
}

// Same, but source is [N][K] row-major (i.e. B = src^T): B[k][n] = s[n][k].
__global__ void k_cvtBT(const float* __restrict__ s, h16* __restrict__ d,
                        int K, int N) {
    size_t tot = (size_t)K * N;
    int nt64 = N >> 6;
    for (size_t i = (size_t)blockIdx.x * blockDim.x + threadIdx.x; i < tot;
         i += (size_t)gridDim.x * blockDim.x) {
        int k = (int)(i / N), nn = (int)(i % N);
        float v = s[(size_t)nn * K + k];
        size_t base = ((size_t)(k >> 5) * nt64 + (nn >> 6)) * 2048;
        d[base + swB_idx(k, nn)] = (h16)v;
    }
}

// Swizzle the V head-slices of qkv into fragment order, per (b,h):
// V is [S keys][64 d]; tiles along keys (S/32 of them), 2048 halves each.
__global__ void k_swzV(const h16* __restrict__ qkv, h16* __restrict__ vsw) {
    size_t tot = (size_t)BB * SS * NHD * DHD;
    for (size_t i = (size_t)blockIdx.x * blockDim.x + threadIdx.x; i < tot;
         i += (size_t)gridDim.x * blockDim.x) {
        int d = (int)(i & 63);
        size_t r = i >> 6;
        int h = (int)(r % NHD);
        size_t rk = r / NHD;
        int key = (int)(rk & (SS - 1));
        int b = (int)(rk >> 9);
        h16 v = qkv[((size_t)b * SS + key) * (3 * DD) + 2 * DD + h * DHD + d];
        size_t base = ((size_t)(b * NHD + h) * (SS / 32) + (key >> 5)) * 2048;
        vsw[base + swB_idx(key, d)] = v;
    }
}

// ---------------------------------------------------------------------------
// block reduction helper (256 threads)
// ---------------------------------------------------------------------------
__device__ __forceinline__ float block_sum256(float v, float* red) {
    int t = threadIdx.x;
    red[t] = v;
    __syncthreads();
#pragma unroll
    for (int st = 128; st > 0; st >>= 1) {
        if (t < st) red[t] += red[t + st];
        __syncthreads();
    }
    float r = red[0];
    __syncthreads();
    return r;
}

// ---------------------------------------------------------------------------
// Embedding gather + LayerNorm. One block per token row (B*S rows).
// ---------------------------------------------------------------------------
__global__ __launch_bounds__(256) void k_embed(const int* __restrict__ ids,
                                               const float* __restrict__ tok,
                                               const float* __restrict__ pos,
                                               const float* __restrict__ typ,
                                               const float* __restrict__ lns,
                                               const float* __restrict__ lnb,
                                               float* __restrict__ xf,
                                               h16* __restrict__ xh) {
    __shared__ float red[256];
    int row = blockIdx.x;
    int s = row & (SS - 1);
    int id = ids[row];
    float loc[3];
    float sum = 0.0f;
#pragma unroll
    for (int j = 0; j < 3; ++j) {
        int d = threadIdx.x + j * 256;
        float v = tok[(size_t)id * DD + d] + pos[(size_t)s * DD + d] + typ[d];
        loc[j] = v;
        sum += v;
    }
    float mu = block_sum256(sum, red) * (1.0f / DD);
    float s2 = 0.0f;
#pragma unroll
    for (int j = 0; j < 3; ++j) {
        float d0 = loc[j] - mu;
        s2 += d0 * d0;
    }
    float var = block_sum256(s2, red) * (1.0f / DD);
    float inv = rsqrtf(var + 1e-12f);
#pragma unroll
    for (int j = 0; j < 3; ++j) {
        int d = threadIdx.x + j * 256;
        float y = (loc[j] - mu) * inv * lns[d] + lnb[d];
        xf[(size_t)row * DD + d] = y;
        xh[(size_t)row * DD + d] = (h16)y;
    }
}

// ---------------------------------------------------------------------------
// Residual add + LayerNorm (in-place on xf), also emits f16 copy.
// ---------------------------------------------------------------------------
__global__ __launch_bounds__(256) void k_addln(float* __restrict__ xf,
                                               const float* __restrict__ tf,
                                               const float* __restrict__ lns,
                                               const float* __restrict__ lnb,
                                               h16* __restrict__ xh) {
    __shared__ float red[256];
    int row = blockIdx.x;
    float loc[3];
    float sum = 0.0f;
#pragma unroll
    for (int j = 0; j < 3; ++j) {
        int d = threadIdx.x + j * 256;
        float v = xf[(size_t)row * DD + d] + tf[(size_t)row * DD + d];
        loc[j] = v;
        sum += v;
    }
    float mu = block_sum256(sum, red) * (1.0f / DD);
    float s2 = 0.0f;
#pragma unroll
    for (int j = 0; j < 3; ++j) {
        float d0 = loc[j] - mu;
        s2 += d0 * d0;
    }
    float var = block_sum256(s2, red) * (1.0f / DD);
    float inv = rsqrtf(var + 1e-12f);
#pragma unroll
    for (int j = 0; j < 3; ++j) {
        int d = threadIdx.x + j * 256;
        float y = (loc[j] - mu) * inv * lns[d] + lnb[d];
        xf[(size_t)row * DD + d] = y;
        xh[(size_t)row * DD + d] = (h16)y;
    }
}

// ---------------------------------------------------------------------------
// WMMA GEMM: C[MxN] = act(A[MxK](f16) * B(swizzled f16) + bias)
// 128 threads = 4 waves; block tile 64x64; wave computes 16x64.
// Staging uses CDNA5 direct-to-LDS paths:
//   B tile (uniform 4KB copy) via TDM tensor_load_to_lds (wave 0, TENSORcnt)
//   A tile via per-lane global_load_async_to_lds_b128 (ASYNCcnt)
// ---------------------------------------------------------------------------
__global__ __launch_bounds__(128) void k_gemm(const h16* __restrict__ A,
                                              const h16* __restrict__ Bs,
                                              const float* __restrict__ bias,
                                              float* __restrict__ Cf,
                                              h16* __restrict__ Ch,
                                              int M, int N, int Np, int K,
                                              int gelu) {
    __shared__ __align__(32) h16 At[64 * 32];
    __shared__ __align__(32) h16 Bt[2048];
    int tid = threadIdx.x;
    int lane = tid & 31;
    int w = tid >> 5;
    int n0 = blockIdx.x * 64;
    int m0 = blockIdx.y * 64;
    int ntiles = Np >> 6;

    v8f acc[4];
#pragma unroll
    for (int c = 0; c < 4; ++c) acc[c] = zero8();

    for (int k0 = 0; k0 < K; k0 += 32) {
        // ---- B tile: TDM DMA of one contiguous 4KB fragment-swizzled tile ----
        if (tid < 32) {
            const h16* bsrc =
                Bs + ((size_t)(k0 >> 5) * ntiles + (n0 >> 6)) * 2048;
            unsigned long long ga = (unsigned long long)(size_t)bsrc;
            u32x4 g0;
            g0[0] = 1u;  // count=1 valid user descriptor; no gather
            g0[1] = lds_off(Bt);                    // lds_addr[31:0]
            g0[2] = (unsigned int)ga;               // global_addr[31:0]
            g0[3] = (unsigned int)((ga >> 32) & 0x01FFFFFFu) |
                    (2u << 30);                     // global_addr[56:32], type=2
            u32x8 g1;
            g1[0] = 1u << 16;                       // data_size = 2 bytes
            g1[1] = (2048u & 0xFFFFu) << 16;        // tensor_dim0[15:0]=2048
            g1[2] = (2048u >> 16) | (1u << 16);     // tensor_dim0[31:16]; dim1=1
            g1[3] = 2048u << 16;                    // tile_dim0 = 2048
            g1[4] = 1u;                             // tile_dim1 = 1
            g1[5] = 2048u;                          // tensor_dim0_stride[31:0]
            g1[6] = 0u;
            g1[7] = 0u;
            asm volatile("tensor_load_to_lds %0, %1"
                         :
                         : "s"(g0), "s"(g1)
                         : "memory");
        }
        // ---- A tile: async DMA, 2 x b128 per thread (offset hits LDS+mem) ----
        {
            int r = tid >> 1, cb = (tid & 1) * 16;
            const h16* gsrc = A + (size_t)(m0 + r) * K + k0 + cb;
            unsigned int la = lds_off(At + r * 32 + cb);
            asm volatile("global_load_async_to_lds_b128 %0, %1, off"
                         :
                         : "v"(la), "v"(gsrc)
                         : "memory");
            asm volatile("global_load_async_to_lds_b128 %0, %1, off offset:16"
                         :
                         : "v"(la), "v"(gsrc)
                         : "memory");
        }
        asm volatile("s_wait_asynccnt 0x0" ::: "memory");
        if (tid < 32) __builtin_amdgcn_s_wait_tensorcnt(0);
        __syncthreads();

        v16h af = frag_a(At + (w * 16) * 32, 32, lane);
#pragma unroll
        for (int c = 0; c < 4; ++c) {
            v16h bf = ldv(Bt + c * 512 + lane * 16);  // one contiguous 32B load
            acc[c] = wmma32(af, bf, acc[c]);
        }
        __syncthreads();
    }

    int lane_n = lane & 15;
    int hf = lane >> 4;
#pragma unroll
    for (int c = 0; c < 4; ++c) {
        int gn = n0 + c * 16 + lane_n;
        if (gn < N) {
            float bv = bias ? bias[gn] : 0.0f;
#pragma unroll
            for (int i = 0; i < 8; ++i) {
                int gm = m0 + w * 16 + hf * 8 + i;
                float v = acc[c][i] + bv;
                if (gelu) v = 0.5f * v * (1.0f + erff(v * 0.70710678f));
                if (Cf) Cf[(size_t)gm * N + gn] = v;
                if (Ch) Ch[(size_t)gm * N + gn] = (h16)v;
            }
        }
    }
}

// ---------------------------------------------------------------------------
// Flash attention with WMMA. grid = (S/64 q-tiles, B*NH). 128 threads = 4 waves.
// K^T fragments: contiguous 32B loads straight from row-major qkv.
// V fragments: contiguous 32B loads from the pre-swizzled vsw buffer.
// Only P (C-frag -> A-frag layout conversion) goes through wave-private LDS.
// ---------------------------------------------------------------------------
__global__ __launch_bounds__(128) void k_attn(const h16* __restrict__ qkv,
                                              const h16* __restrict__ vsw,
                                              const int* __restrict__ mask,
                                              h16* __restrict__ out) {
    __shared__ __align__(32) h16 Pt[4][16 * 64];
    int tid = threadIdx.x, lane = tid & 31, w = tid >> 5;
    int bh = blockIdx.y;
    int b = bh / NHD, hh = bh % NHD;
    int q0 = blockIdx.x * 64;
    size_t row0 = (size_t)b * SS;
    int qbase = q0 + w * 16;

    // Q fragments (held in registers for the whole key loop)
    v16h aq[2];
    {
        int r = lane & 15, kb = (lane >> 4) * 16;
        const h16* qp = qkv + (row0 + qbase + r) * (3 * DD) + hh * DHD;
#pragma unroll
        for (int ch = 0; ch < 2; ++ch) {
#pragma unroll
            for (int j = 0; j < 8; ++j) aq[ch][j] = qp[ch * 32 + kb + j];
#pragma unroll
            for (int j = 0; j < 8; ++j) aq[ch][8 + j] = qp[ch * 32 + kb + 8 + j];
        }
    }

    v8f oacc[4];
#pragma unroll
    for (int c = 0; c < 4; ++c) oacc[c] = zero8();
    float mrow[8], lrow[8];
#pragma unroll
    for (int i = 0; i < 8; ++i) {
        mrow[i] = -1e30f;
        lrow[i] = 0.0f;
    }

    size_t vbh = (size_t)(b * NHD + hh) * (SS / 32) * 2048;
    int kchunk = (lane >> 4) * 16;

    for (int kb0 = 0; kb0 < SS; kb0 += 64) {
        // scores tile 16x64 (this wave's rows)
        v8f s[4];
#pragma unroll
        for (int c = 0; c < 4; ++c) {
            int keyrow = kb0 + c * 16 + (lane & 15);
            const h16* kp =
                qkv + (row0 + keyrow) * (3 * DD) + DD + hh * DHD + kchunk;
            v8f sc = zero8();
            sc = wmma32(aq[0], ldv(kp), sc);
            sc = wmma32(aq[1], ldv(kp + 32), sc);
            int key = kb0 + c * 16 + (lane & 15);
            float ng = (1.0f - (float)mask[row0 + key]) * -1e9f;
#pragma unroll
            for (int i = 0; i < 8; ++i) s[c][i] = sc[i] * 0.125f + ng;
        }

        // online softmax (row reductions over 16-lane half-groups)
        float pfac[8];
#pragma unroll
        for (int i = 0; i < 8; ++i) {
            float v = fmaxf(fmaxf(s[0][i], s[1][i]), fmaxf(s[2][i], s[3][i]));
            v = fmaxf(v, __shfl_xor(v, 1, 16));
            v = fmaxf(v, __shfl_xor(v, 2, 16));
            v = fmaxf(v, __shfl_xor(v, 4, 16));
            v = fmaxf(v, __shfl_xor(v, 8, 16));
            float mnew = fmaxf(mrow[i], v);
            pfac[i] = expf(mrow[i] - mnew);
            mrow[i] = mnew;
            lrow[i] *= pfac[i];
        }
#pragma unroll
        for (int c = 0; c < 4; ++c)
#pragma unroll
            for (int i = 0; i < 8; ++i) oacc[c][i] *= pfac[i];

        float rsum[8];
#pragma unroll
        for (int i = 0; i < 8; ++i) rsum[i] = 0.0f;
#pragma unroll
        for (int c = 0; c < 4; ++c)
#pragma unroll
            for (int i = 0; i < 8; ++i) {
                float p = expf(s[c][i] - mrow[i]);
                s[c][i] = p;
                rsum[i] += p;
            }
#pragma unroll
        for (int i = 0; i < 8; ++i) {
            float v = rsum[i];
            v += __shfl_xor(v, 1, 16);
            v += __shfl_xor(v, 2, 16);
            v += __shfl_xor(v, 4, 16);
            v += __shfl_xor(v, 8, 16);
            lrow[i] += v;
        }

        // stage P (f16) through wave-private LDS to re-enter as A-fragment
        {
            h16* pw = &Pt[w][0];
            int n = lane & 15, hf = lane >> 4;
#pragma unroll
            for (int c = 0; c < 4; ++c)
#pragma unroll
                for (int i = 0; i < 8; ++i)
                    pw[(hf * 8 + i) * 64 + c * 16 + n] = (h16)s[c][i];
        }
        v16h pa0 = frag_a(&Pt[w][0], 64, lane);
        v16h pa1 = frag_a(&Pt[w][0] + 32, 64, lane);

        const h16* vp = vsw + vbh + (size_t)(kb0 >> 5) * 2048;
#pragma unroll
        for (int c = 0; c < 4; ++c) {
            v16h bv0 = ldv(vp + c * 512 + lane * 16);
            v16h bv1 = ldv(vp + 2048 + c * 512 + lane * 16);
            oacc[c] = wmma32(pa0, bv0, oacc[c]);
            oacc[c] = wmma32(pa1, bv1, oacc[c]);
        }
    }

    // normalize + store
    int n = lane & 15, hf = lane >> 4;
#pragma unroll
    for (int c = 0; c < 4; ++c)
#pragma unroll
        for (int i = 0; i < 8; ++i) {
            size_t r = row0 + qbase + hf * 8 + i;
            out[r * DD + hh * DHD + c * 16 + n] = (h16)(oacc[c][i] / lrow[i]);
        }
}

// ---------------------------------------------------------------------------
// cumsum of n_pieces per batch (tiny)
// ---------------------------------------------------------------------------
__global__ void k_cumsum(const int* __restrict__ np, int* __restrict__ cum) {
    int b = blockIdx.x;
    if (threadIdx.x == 0) {
        int run = 0;
        cum[b * (WW + 1)] = 0;
        for (int w = 0; w < WW; ++w) {
            run += np[b * WW + w];
            cum[b * (WW + 1) + w + 1] = run;
        }
    }
}

// ---------------------------------------------------------------------------
// subword mean pool + passthrough tail; emits f16 LSTM input [B*S][D]
// ---------------------------------------------------------------------------
__global__ __launch_bounds__(256) void k_pool(const float* __restrict__ xf,
                                              const int* __restrict__ np,
                                              const int* __restrict__ cum,
                                              h16* __restrict__ outp) {
    int row = blockIdx.x;
    int b = row >> 9, s = row & (SS - 1);
    if (s < WW) {
        int start = cum[b * (WW + 1) + s];
        int cnt = np[b * WW + s];
        float invc = 1.0f / (float)cnt;
#pragma unroll
        for (int j = 0; j < 3; ++j) {
            int d = threadIdx.x + j * 256;
            float acc = 0.0f;
            for (int q = 0; q < cnt; ++q)
                acc += xf[((size_t)b * SS + start + q) * DD + d];
            outp[(size_t)row * DD + d] = (h16)(acc * invc);
        }
    } else {
#pragma unroll
        for (int j = 0; j < 3; ++j) {
            int d = threadIdx.x + j * 256;
            outp[(size_t)row * DD + d] = (h16)xf[(size_t)row * DD + d];
        }
    }
}

// ---------------------------------------------------------------------------
// Persistent BiLSTM recurrence. grid=2 (dir), 768 threads = 24 waves.
// Each wave owns a 16-wide hidden slice u0 across all 4 gates, so i/f/g/o land
// element-aligned in the same C-fragment lanes; cell state stays in registers.
// xi: precomputed x@Wih^T + b, [B*S][1536] f32. WhhT: swizzled-B, K=384 N=1536.
// f1: [B*S][768] f16 ([hf | hb]).
// ---------------------------------------------------------------------------
__global__ __launch_bounds__(768) void k_lstm(const float* __restrict__ xiF,
                                              const float* __restrict__ xiB,
                                              const h16* __restrict__ WhhTf,
                                              const h16* __restrict__ WhhTb,
                                              h16* __restrict__ f1) {
    __shared__ __align__(32) h16 hsh[16 * LH];
    int tid = threadIdx.x, lane = tid & 31, w = tid >> 5;
    int dir = blockIdx.x;
    const float* xi = dir ? xiB : xiF;
    const h16* Wt = dir ? WhhTb : WhhTf;
    for (int i = tid; i < 16 * LH; i += 768) hsh[i] = (h16)0.0f;
    float cst[8];
#pragma unroll
    for (int i = 0; i < 8; ++i) cst[i] = 0.0f;
    int u0 = w * 16;
    int n = lane & 15, hf = lane >> 4;
    __syncthreads();

    for (int t = 0; t < SS; ++t) {
        int tt = dir ? (SS - 1 - t) : t;
        v8f acc[4];
#pragma unroll
        for (int g = 0; g < 4; ++g) {
#pragma unroll
            for (int i = 0; i < 8; ++i)
                acc[g][i] = (hf == 0)
                                ? xi[((size_t)i * SS + tt) * G4 + g * LH + u0 + n]
                                : 0.0f;
        }
#pragma unroll 4
        for (int kc = 0; kc < 12; ++kc) {
            v16h ah = frag_a(hsh + kc * 32, LH, lane);
#pragma unroll
            for (int g = 0; g < 4; ++g) {
                int gcol = g * LH + u0;  // column base for this gate's slice
                const h16* wp = Wt + ((size_t)kc * (G4 >> 6) + (gcol >> 6)) * 2048 +
                                ((gcol & 63) >> 4) * 512 + lane * 16;
                acc[g] = wmma32(ah, ldv(wp), acc[g]);
            }
        }
        __syncthreads();  // all hsh reads done before overwrite
#pragma unroll
        for (int i = 0; i < 8; ++i) {
            float iv = sigf(acc[0][i]);
            float fv = sigf(acc[1][i]);
            float gv = tanhf(acc[2][i]);
            float ov = sigf(acc[3][i]);
            float c = fv * cst[i] + iv * gv;
            cst[i] = c;
            float hval = ov * tanhf(c);
            h16 hh = (h16)hval;
            hsh[(hf * 8 + i) * LH + u0 + n] = hh;
            if (hf == 0)
                f1[((size_t)i * SS + tt) * (2 * LH) + dir * LH + u0 + n] = hh;
        }
        __syncthreads();
    }
}

// ---------------------------------------------------------------------------
// host orchestration
// ---------------------------------------------------------------------------
extern "C" void kernel_launch(void* const* d_in, const int* in_sizes, int n_in,
                              void* d_out, int out_size, void* d_ws, size_t ws_size,
                              hipStream_t stream) {
    (void)in_sizes; (void)n_in; (void)out_size; (void)ws_size;
    // inputs: recursive insertion-order flatten of setup_inputs()
    const int*   enc  = (const int*)d_in[0];
    const int*   msk  = (const int*)d_in[1];
    const int*   npc  = (const int*)d_in[2];
    const float* tok  = (const float*)d_in[3];
    const float* pos  = (const float*)d_in[4];
    const float* typ  = (const float*)d_in[5];
    const float* elns = (const float*)d_in[6];
    const float* elnb = (const float*)d_in[7];
    const float* Wqkv = (const float*)d_in[8];
    const float* bqkv = (const float*)d_in[9];
    const float* Wo   = (const float*)d_in[10];
    const float* bo   = (const float*)d_in[11];
    const float* ln1s = (const float*)d_in[12];
    const float* ln1b = (const float*)d_in[13];
    const float* W1   = (const float*)d_in[14];
    const float* b1   = (const float*)d_in[15];
    const float* W2   = (const float*)d_in[16];
    const float* b2   = (const float*)d_in[17];
    const float* ln2s = (const float*)d_in[18];
    const float* ln2b = (const float*)d_in[19];
    const float* Wihf = (const float*)d_in[20];
    const float* Whhf = (const float*)d_in[21];
    const float* bfwd = (const float*)d_in[22];
    const float* Wihb = (const float*)d_in[23];
    const float* Whhb = (const float*)d_in[24];
    const float* bbwd = (const float*)d_in[25];
    const float* Wout = (const float*)d_in[26];
    const float* bout = (const float*)d_in[27];
    float* out = (float*)d_out;

    // carve workspace
    size_t off = 0;
    char* base = (char*)d_ws;
    auto carve = [&](size_t bytes) -> void* {
        void* p = base + off;
        off = (off + bytes + 255) & ~(size_t)255;
        return p;
    };
    h16* Wqkv_h = (h16*)carve((size_t)LL * DD * 3 * DD * 2);
    h16* Wo_h   = (h16*)carve((size_t)LL * DD * DD * 2);
    h16* W1_h   = (h16*)carve((size_t)LL * DD * FF * 2);
    h16* W2_h   = (h16*)carve((size_t)LL * FF * DD * 2);
    h16* Wout_h = (h16*)carve((size_t)(2 * LH) * NCP * 2);
    h16* WihT_f = (h16*)carve((size_t)DD * G4 * 2);
    h16* WihT_b = (h16*)carve((size_t)DD * G4 * 2);
    h16* WhhT_f = (h16*)carve((size_t)LH * G4 * 2);
    h16* WhhT_b = (h16*)carve((size_t)LH * G4 * 2);
    h16* xh     = (h16*)carve((size_t)ROWS * DD * 2);
    h16* qkv_h  = (h16*)carve((size_t)ROWS * 3 * DD * 2);
    h16* vsw    = (h16*)carve((size_t)BB * NHD * (SS / 32) * 2048 * 2);
    h16* ao_h   = (h16*)carve((size_t)ROWS * DD * 2);
    h16* ff_h   = (h16*)carve((size_t)ROWS * FF * 2);
    h16* pool_h = (h16*)carve((size_t)ROWS * DD * 2);
    h16* f1_h   = (h16*)carve((size_t)ROWS * 2 * LH * 2);
    float* x_f  = (float*)carve((size_t)ROWS * DD * 4);
    float* tmpf = (float*)carve((size_t)ROWS * DD * 4);
    float* xi_f = (float*)carve((size_t)ROWS * G4 * 4);
    float* xi_b = (float*)carve((size_t)ROWS * G4 * 4);
    int* cum    = (int*)carve((size_t)BB * (WW + 1) * 4);

    // weight conversion into fragment-swizzled f16 (deterministic each call)
    k_cvtB<<<2048, 256, 0, stream>>>(Wqkv, Wqkv_h, LL * DD, 3 * DD, 3 * DD);
    k_cvtB<<<2048, 256, 0, stream>>>(Wo,   Wo_h,   LL * DD, DD, DD);
    k_cvtB<<<2048, 256, 0, stream>>>(W1,   W1_h,   LL * DD, FF, FF);
    k_cvtB<<<2048, 256, 0, stream>>>(W2,   W2_h,   LL * FF, DD, DD);
    k_cvtB<<<512,  256, 0, stream>>>(Wout, Wout_h, 2 * LH, NC, NCP);
    k_cvtBT<<<1024, 256, 0, stream>>>(Wihf, WihT_f, DD, G4);
    k_cvtBT<<<1024, 256, 0, stream>>>(Wihb, WihT_b, DD, G4);
    k_cvtBT<<<1024, 256, 0, stream>>>(Whhf, WhhT_f, LH, G4);
    k_cvtBT<<<1024, 256, 0, stream>>>(Whhb, WhhT_b, LH, G4);
    // NOTE: per-layer weight matrices are stacked along K (rows), and K-tiling
    // is 32-aligned; since DD and FF are multiples of 32, tile rows never
    // straddle layers, so converting [L*K][N] in one pass is layout-identical
    // to converting each layer separately.

    // embeddings + LN
    k_embed<<<ROWS, 256, 0, stream>>>(enc, tok, pos, typ, elns, elnb, x_f, xh);

    // BERT encoder
    for (int l = 0; l < LL; ++l) {
        k_gemm<<<dim3((3 * DD) / 64, ROWS / 64), 128, 0, stream>>>(
            xh, Wqkv_h + (size_t)l * DD * 3 * DD, bqkv + (size_t)l * 3 * DD,
            nullptr, qkv_h, ROWS, 3 * DD, 3 * DD, DD, 0);
        k_swzV<<<1024, 256, 0, stream>>>(qkv_h, vsw);
        k_attn<<<dim3(SS / 64, BB * NHD), 128, 0, stream>>>(qkv_h, vsw, msk, ao_h);
        k_gemm<<<dim3(DD / 64, ROWS / 64), 128, 0, stream>>>(
            ao_h, Wo_h + (size_t)l * DD * DD, bo + (size_t)l * DD,
            tmpf, nullptr, ROWS, DD, DD, DD, 0);
        k_addln<<<ROWS, 256, 0, stream>>>(x_f, tmpf, ln1s + (size_t)l * DD,
                                          ln1b + (size_t)l * DD, xh);
        k_gemm<<<dim3(FF / 64, ROWS / 64), 128, 0, stream>>>(
            xh, W1_h + (size_t)l * DD * FF, b1 + (size_t)l * FF,
            nullptr, ff_h, ROWS, FF, FF, DD, 1);
        k_gemm<<<dim3(DD / 64, ROWS / 64), 128, 0, stream>>>(
            ff_h, W2_h + (size_t)l * FF * DD, b2 + (size_t)l * DD,
            tmpf, nullptr, ROWS, DD, DD, FF, 0);
        k_addln<<<ROWS, 256, 0, stream>>>(x_f, tmpf, ln2s + (size_t)l * DD,
                                          ln2b + (size_t)l * DD, xh);
    }

    // subword mean pool
    k_cumsum<<<BB, 32, 0, stream>>>(npc, cum);
    k_pool<<<ROWS, 256, 0, stream>>>(x_f, npc, cum, pool_h);

    // LSTM input projections (bias folded in)
    k_gemm<<<dim3(G4 / 64, ROWS / 64), 128, 0, stream>>>(
        pool_h, WihT_f, bfwd, xi_f, nullptr, ROWS, G4, G4, DD, 0);
    k_gemm<<<dim3(G4 / 64, ROWS / 64), 128, 0, stream>>>(
        pool_h, WihT_b, bbwd, xi_b, nullptr, ROWS, G4, G4, DD, 0);

    // BiLSTM recurrence
    k_lstm<<<2, 768, 0, stream>>>(xi_f, xi_b, WhhT_f, WhhT_b, f1_h);

    // output projection (B padded to 448 cols; C writes guarded by N=425)
    k_gemm<<<dim3(NCP / 64, ROWS / 64), 128, 0, stream>>>(
        f1_h, Wout_h, bout, out, nullptr, ROWS, NC, NCP, 2 * LH, 0);
}
